// BiLSTM_CRF_18098992185950
// MI455X (gfx1250) — matrix operand order repaired and checked by reading, not compiled, not run on hardware
//
#include <hip/hip_runtime.h>
#include <hip/hip_bf16.h>

typedef __attribute__((ext_vector_type(16))) _Float16 v16h;
typedef __attribute__((ext_vector_type(8)))  float    v8f;

#define D_EMB 256
#define TF    1024
#define KC    64
#define NROWS (TF * KC)   // 65536
#define NCHUNK 8          // D_EMB / 32 K-chunks

// ---------------------------------------------------------------------------
// Kernel 0: collapse conv+linear into 4 effective weight vectors + 2 consts,
// then pre-bake the constant WMMA B-operand (D x 16, cols 0..2 = weff_e1/
// weff_t0/weff_t1, cols 3..15 = 0) into f16 *fragment layout*:
//   Bh[((ch*32 + lane)*16) + e]  with  K = ch*32 + (lane>>4)*16 + e,
//                                      N = lane & 15.
// So crf_proj loads each lane's fragment with one aligned 32B vector load.
// ---------------------------------------------------------------------------
__global__ void crf_prep(const float* __restrict__ ecw, const float* __restrict__ ecb,
                         const float* __restrict__ elw, const float* __restrict__ elb,
                         const float* __restrict__ tcw, const float* __restrict__ tcb,
                         const float* __restrict__ tlw, const float* __restrict__ tlb,
                         float* __restrict__ W, float* __restrict__ C,
                         _Float16* __restrict__ Bh) {
  int d = threadIdx.x;                       // 256 threads
  float eLm1 = (d > 0)   ? elw[d - 1] : 0.f;
  float eL   = elw[d];
  float eLp1 = (d < 255) ? elw[d + 1] : 0.f;
  float tLm1 = (d > 0)   ? tlw[d - 1] : 0.f;
  float tL   = tlw[d];
  float tLp1 = (d < 255) ? tlw[d + 1] : 0.f;
  W[d]       = ecw[0] * eLp1 + ecw[1] * eL + ecw[2] * eLm1;   // feat -> emit
  W[256 + d] = ecw[3] * eLp1 + ecw[4] * eL + ecw[5] * eLm1;   // cand -> emit
  W[512 + d] = tcw[0] * tLp1 + tcw[1] * tL + tcw[2] * tLm1;   // cand -> u (prev)
  W[768 + d] = tcw[3] * tLp1 + tcw[4] * tL + tcw[5] * tLm1;   // cand -> v (cur)
  if (d == 0) {
    float se = 0.f, st = 0.f;
    for (int i = 0; i < D_EMB; ++i) { se += elw[i]; st += tlw[i]; }
    C[0] = ecb[0] * se + elb[0];
    C[1] = tcb[0] * st + tlb[0];
  }
  __syncthreads();
  // Build the constant f16 B fragments: 8 chunks x 32 lanes, 16 halfs each.
  const int slot = threadIdx.x;              // == ch*32 + lane
  const int ch   = slot >> 5;
  const int lane = slot & 31;
  const int n    = lane & 15;
  const int kb2  = (lane >> 4) * 16;
#pragma unroll
  for (int e = 0; e < 16; ++e) {
    const int K = ch * 32 + kb2 + e;
    const float bv = (n < 3) ? W[256 * (n + 1) + K] : 0.f;
    Bh[slot * 16 + e] = (_Float16)bv;
  }
}

// ---------------------------------------------------------------------------
// Kernel 1: gather + skinny GEMM via WMMA f32_16x16x32_f16.
// One wave per 16-row tile; 8 K=32 chunks over D=256.  A rows are gathered
// embedding rows (f32 -> f16 pack-convert); B fragments come pre-baked from
// Bh with a single 32B load per lane per chunk (branch-free hot loop).
// C-fragment columns 0/1/2 are scattered to EC/U/V.
// ---------------------------------------------------------------------------
__global__ void crf_proj(const int* __restrict__ cids, const float* __restrict__ wemb,
                         const _Float16* __restrict__ Bh,
                         float* __restrict__ EC, float* __restrict__ U,
                         float* __restrict__ V) {
  const int lane = threadIdx.x & 31;
  const int wave = threadIdx.x >> 5;
  const int tile = blockIdx.x * 8 + wave;    // 4096 tiles of 16 rows
  const int n    = lane & 15;                // A-row index AND C/D column index
  const int half = lane >> 4;

  const int arow = tile * 16 + n;
  const int cid  = cids[arow];
  const float* rp = wemb + (size_t)cid * D_EMB;
  const v16h* Bf = reinterpret_cast<const v16h*>(Bh);

  const int kbase = half * 16;               // A: K offsets per half-wave

  v8f c = {};
#pragma unroll
  for (int ch = 0; ch < NCHUNK; ++ch) {
    const int d0 = ch * 32;
    // A fragment: elems 0..15 -> K = d0+kbase..+15
    const float4* pa0 = reinterpret_cast<const float4*>(rp + d0 + kbase);
    const float4* pa1 = reinterpret_cast<const float4*>(rp + d0 + kbase + 8);
    float4 x0 = pa0[0], x1 = pa0[1];
    float4 y0 = pa1[0], y1 = pa1[1];
    v16h a;
    a[0]  = (_Float16)x0.x; a[1]  = (_Float16)x0.y; a[2]  = (_Float16)x0.z; a[3]  = (_Float16)x0.w;
    a[4]  = (_Float16)x1.x; a[5]  = (_Float16)x1.y; a[6]  = (_Float16)x1.z; a[7]  = (_Float16)x1.w;
    a[8]  = (_Float16)y0.x; a[9]  = (_Float16)y0.y; a[10] = (_Float16)y0.z; a[11] = (_Float16)y0.w;
    a[12] = (_Float16)y1.x; a[13] = (_Float16)y1.y; a[14] = (_Float16)y1.z; a[15] = (_Float16)y1.w;
    // B fragment: one aligned 32-byte load, already in fragment layout.
    v16h b = Bf[ch * 32 + lane];
    c = __builtin_amdgcn_wmma_f32_16x16x32_f16(false, a, false, b, (short)0, c,
                                               false, false);
  }
  // C/D layout: column = n, row M = half*8 + r
  const int mbase = tile * 16 + half * 8;
#pragma unroll
  for (int r = 0; r < 8; ++r) {
    const int orow = mbase + r;
    const float val = c[r];
    if      (n == 0) EC[orow] = val;
    else if (n == 1) U[orow]  = val;
    else if (n == 2) V[orow]  = val;
  }
}

// ---------------------------------------------------------------------------
// Kernel 2: FE[t] = feat_t . weff_e0   (one wave per row, shuffle reduce)
// ---------------------------------------------------------------------------
__global__ void crf_feat(const float* __restrict__ feats, const float* __restrict__ W,
                         float* __restrict__ FE) {
  const int lane = threadIdx.x & 31;
  const int wave = threadIdx.x >> 5;
  const int t = blockIdx.x * 8 + wave;
  const float4* p = reinterpret_cast<const float4*>(feats + (size_t)t * D_EMB + lane * 8);
  const float4* q = reinterpret_cast<const float4*>(W + lane * 8);
  float4 a0 = p[0], a1 = p[1], b0 = q[0], b1 = q[1];
  float acc = a0.x * b0.x + a0.y * b0.y + a0.z * b0.z + a0.w * b0.w
            + a1.x * b1.x + a1.y * b1.y + a1.z * b1.z + a1.w * b1.w;
#pragma unroll
  for (int off = 16; off > 0; off >>= 1) acc += __shfl_down(acc, off, 32);
  if (lane == 0) FE[t] = acc;
}

// ---------------------------------------------------------------------------
// Kernel 3: sequential CRF forward DP (1 block, 64 threads; thread k owns
// alpha[k]).  Stable logsumexp uses m = max(alpha)+1 since sigmoid in (0,1).
// ---------------------------------------------------------------------------
__device__ __forceinline__ float sigmoidf_(float x) {
  return 1.f / (1.f + __expf(-x));
}

__global__ void crf_dp(const float* __restrict__ EC, const float* __restrict__ U,
                       const float* __restrict__ V, const float* __restrict__ FE,
                       const float* __restrict__ C, float* __restrict__ out) {
  __shared__ float alpha[KC];
  __shared__ float ush[KC];
  __shared__ float red[KC];
  const int k = threadIdx.x;                 // 64 threads
  const float ce = C[0], ct = C[1];

  alpha[k] = sigmoidf_(FE[0] + EC[k] + ce);  // alpha_0 = emit[0]
  __syncthreads();

  for (int t = 1; t < TF; ++t) {
    red[k] = alpha[k];
    ush[k] = U[(t - 1) * KC + k];
    __syncthreads();
#pragma unroll
    for (int off = 32; off > 0; off >>= 1) {
      if (k < off) red[k] = fmaxf(red[k], red[k + off]);
      __syncthreads();
    }
    const float m  = red[0] + 1.f;           // >= alpha[j] + sigmoid(.) for all j
    const float vk = V[t * KC + k];
    float acc = 0.f;
#pragma unroll 8
    for (int j = 0; j < KC; ++j)
      acc += __expf(alpha[j] + sigmoidf_(ush[j] + vk + ct) - m);
    const float e  = sigmoidf_(FE[t] + EC[t * KC + k] + ce);
    const float na = m + __logf(acc) + e;
    __syncthreads();
    alpha[k] = na;
    __syncthreads();
  }

  // final logsumexp over alpha
  red[k] = alpha[k];
  __syncthreads();
#pragma unroll
  for (int off = 32; off > 0; off >>= 1) {
    if (k < off) red[k] = fmaxf(red[k], red[k + off]);
    __syncthreads();
  }
  const float m = red[0];
  red[k] = __expf(alpha[k] - m);
  __syncthreads();
#pragma unroll
  for (int off = 32; off > 0; off >>= 1) {
    if (k < off) red[k] += red[k + off];
    __syncthreads();
  }
  if (k == 0) out[0] = m + __logf(red[0]);
}

// ---------------------------------------------------------------------------
extern "C" void kernel_launch(void* const* d_in, const int* in_sizes, int n_in,
                              void* d_out, int out_size, void* d_ws, size_t ws_size,
                              hipStream_t stream) {
  const float* feats = (const float*)d_in[0];   // (TF, D)
  const int*   cids  = (const int*)  d_in[1];   // (TF, K) int32
  const float* wemb  = (const float*)d_in[2];   // (V, D)
  const float* ecw   = (const float*)d_in[3];   // (1,2,3)
  const float* ecb   = (const float*)d_in[4];   // (1,)
  const float* elw   = (const float*)d_in[5];   // (1,D)
  const float* elb   = (const float*)d_in[6];   // (1,)
  const float* tcw   = (const float*)d_in[7];   // (1,2,3)
  const float* tcb   = (const float*)d_in[8];   // (1,)
  const float* tlw   = (const float*)d_in[9];   // (1,D)
  const float* tlb   = (const float*)d_in[10];  // (1,)

  float* ws = (float*)d_ws;
  float* W  = ws;                 // 1024 floats: 4 effective weight vectors
  float* C  = ws + 1024;          // 2 consts (pad to 8)
  float* FE = ws + 1032;          // 1024
  float* EC = ws + 2056;          // 65536
  float* U  = EC + NROWS;         // 65536
  float* V  = U + NROWS;          // 65536
  _Float16* Bh = (_Float16*)(V + NROWS);  // 4096 halfs (8KB), 32B-aligned
                                          // total ws use: ~803 KB

  crf_prep<<<1, 256, 0, stream>>>(ecw, ecb, elw, elb, tcw, tcb, tlw, tlb, W, C, Bh);
  crf_proj<<<NROWS / 16 / 8, 256, 0, stream>>>(cids, wemb, Bh, EC, U, V);
  crf_feat<<<TF / 8, 256, 0, stream>>>(feats, W, FE);
  crf_dp<<<1, KC, 0, stream>>>(EC, U, V, FE, C, (float*)d_out);
}
